// GAT_23991687316175
// MI455X (gfx1250) — matrix operand order repaired and checked by reading, not compiled, not run on hardware
//
#include <hip/hip_runtime.h>
#include <hip/hip_bf16.h>
#include <math.h>

// Problem structural constants (fixed by the reference model)
#define GAT_FIN   128
#define GAT_NCLS  16
#define NEG_SLOPE 0.2f

typedef __attribute__((ext_vector_type(16))) __bf16 v16bf;
typedef __attribute__((ext_vector_type(8)))  float  v8f;

union BfFrag { v16bf v; unsigned short u[16]; uint4 q[2]; };

// ---------- helpers ----------
__device__ inline unsigned short f32_to_bf16_rne(float f) {
  union { float f; unsigned u; } cv; cv.f = f;
  unsigned u = cv.u;
  if ((u & 0x7F800000u) != 0x7F800000u)           // keep inf/nan bits
    u += 0x7FFFu + ((u >> 16) & 1u);              // round-to-nearest-even
  return (unsigned short)(u >> 16);
}

// Monotone float<->uint encoding so atomicMax(u32) orders like float compare.
__device__ inline unsigned enc_f32(float f) {
  union { float f; unsigned u; } cv; cv.f = f;
  return (cv.u & 0x80000000u) ? ~cv.u : (cv.u | 0x80000000u);
}
__device__ inline float dec_f32(unsigned u) {
  union { unsigned u; float f; } cv;
  cv.u = (u & 0x80000000u) ? (u ^ 0x80000000u) : ~u;
  return cv.f;
}

__device__ inline float lrelu(float v) { return v > 0.f ? v : NEG_SLOPE * v; }

// ---------- small elementwise kernels ----------
__global__ void f32_to_bf16_kernel(const float* __restrict__ in,
                                   unsigned short* __restrict__ out, int n) {
  int t = blockIdx.x * blockDim.x + threadIdx.x;
  if (t < n) out[t] = f32_to_bf16_rne(in[t]);
}

__global__ void elu_to_bf16_kernel(const float* __restrict__ in,
                                   unsigned short* __restrict__ out, int n) {
  int t = blockIdx.x * blockDim.x + threadIdx.x;
  if (t < n) {
    float v = in[t];
    v = v > 0.f ? v : (expf(v) - 1.f);
    out[t] = f32_to_bf16_rne(v);
  }
}

__global__ void fill_u32_kernel(unsigned* __restrict__ p, unsigned v, int n) {
  int t = blockIdx.x * blockDim.x + threadIdx.x;
  if (t < n) p[t] = v;
}

__global__ void bias_rows_kernel(float* __restrict__ out,
                                 const float* __restrict__ b, int total, int cols) {
  int t = blockIdx.x * blockDim.x + threadIdx.x;
  if (t < total) out[t] = b[t % cols];
}

// ---------- B pre-swizzle: f32 [128][NCOLS] -> bf16 WMMA-fragment-major ----------
// Layout: Bsw[(((nt*4 + kt)*32 + lane)*16 + e)] = bf16(B[k][n]) with
//   k = kt*32 + (lane>=16 ? 16 : 0) + e,  n = nt*16 + (lane&15)
// so each lane's 16 B elements for one 16x16x32 WMMA are 32 contiguous bytes.
template<int NCOLS>
__global__ void swizzle_B_kernel(const float* __restrict__ B,
                                 unsigned short* __restrict__ Bsw) {
  int t = blockIdx.x * blockDim.x + threadIdx.x;
  if (t >= 128 * NCOLS) return;
  int e    = t & 15;
  int lane = (t >> 4) & 31;
  int kt   = (t >> 9) & 3;
  int nt   = t >> 11;
  int k = kt * 32 + (lane >> 4) * 16 + e;
  int n = nt * 16 + (lane & 15);
  Bsw[t] = f32_to_bf16_rne(B[k * NCOLS + n]);
}

// ---------- bf16 WMMA GEMM: C[M][NCOLS] = A[M][128] * B[128][NCOLS] ----------
// A staged in LDS (fragment runs are contiguous -> ds_load_b128);
// B read as pre-swizzled 128-bit global loads (tiny, L2-resident).
// Two independent accumulator chains per n-tile hide WMMA->WMMA hazards.
template<int NCOLS>
__global__ __launch_bounds__(256)
void gemm_bf16_wmma(const unsigned short* __restrict__ A,     // bf16 [M][128]
                    const unsigned short* __restrict__ Bsw,   // swizzled fragments
                    float* __restrict__ Cout, int M) {
  __shared__ alignas(16) unsigned short Abuf[128 * 128];
  const int tid = threadIdx.x;
  const int m0  = blockIdx.x * 128;
  const bool fullBlock = (m0 + 128 <= M);

  {  // coalesced stage of A tile to LDS (u32 = packed bf16 pair)
    const unsigned* A32 = (const unsigned*)A;
    unsigned* S32 = (unsigned*)Abuf;
    if (fullBlock) {
      for (int i = tid; i < 128 * 64; i += 256)
        S32[i] = A32[(size_t)m0 * 64 + i];
    } else {
      for (int i = tid; i < 128 * 64; i += 256) {
        int gr = m0 + (i >> 6);
        S32[i] = (gr < M) ? A32[(size_t)gr * 64 + (i & 63)] : 0u;
      }
    }
  }
  __syncthreads();

  const int wave = tid >> 5;
  const int lane = tid & 31;
  const int nloc = lane & 15;
  const int hi   = lane >> 4;      // 0 or 1

  // A fragments for this wave's 16 rows, all 4 k-tiles (contiguous k = 16*hi + e, matching Bsw)
  BfFrag afrag[4];
  const int arow = wave * 16 + nloc;   // M = lane%16 in both lane halves
  const int kbA  = hi * 8;
  #pragma unroll
  for (int kt = 0; kt < 4; ++kt) {
    #pragma unroll
    for (int e = 0; e < 16; ++e) {
      int k = kt * 32 + hi * 16 + e;
      afrag[kt].u[e] = Abuf[arow * 128 + k];
    }
  }

  const int grBase = m0 + wave * 16 + hi * 8;

  #pragma unroll
  for (int nt = 0; nt < NCOLS / 16; ++nt) {
    BfFrag bf[4];
    #pragma unroll
    for (int kt = 0; kt < 4; ++kt) {
      const uint4* src = (const uint4*)(Bsw + ((size_t)((nt * 4 + kt) * 32 + lane)) * 16);
      bf[kt].q[0] = src[0];
      bf[kt].q[1] = src[1];
    }
    // two interleaved accumulator chains (kt 0,2 and kt 1,3)
    v8f acc0 = {}, acc1 = {};
    acc0 = __builtin_amdgcn_wmma_f32_16x16x32_bf16(false, afrag[0].v, false, bf[0].v,
                                                   (short)0, acc0, false, false);
    acc1 = __builtin_amdgcn_wmma_f32_16x16x32_bf16(false, afrag[1].v, false, bf[1].v,
                                                   (short)0, acc1, false, false);
    acc0 = __builtin_amdgcn_wmma_f32_16x16x32_bf16(false, afrag[2].v, false, bf[2].v,
                                                   (short)0, acc0, false, false);
    acc1 = __builtin_amdgcn_wmma_f32_16x16x32_bf16(false, afrag[3].v, false, bf[3].v,
                                                   (short)0, acc1, false, false);
    v8f acc = acc0 + acc1;

    // C/D layout: VGPR r -> M = grBase + r, lane%16 -> N.
    // Fast path: one base pointer, compile-time immediate offsets r*NCOLS*4.
    float* cptr = Cout + (size_t)grBase * NCOLS + nt * 16 + nloc;
    if (fullBlock) {
      #pragma unroll
      for (int r = 0; r < 8; ++r) cptr[r * NCOLS] = acc[r];
    } else {
      #pragma unroll
      for (int r = 0; r < 8; ++r)
        if (grBase + r < M) cptr[r * NCOLS] = acc[r];
    }
  }
}

// ---------- attention coefficient projection: alpha[n,h] = <h[n,h,:], att[h,:]> ----------
__global__ void alpha_kernel(const float* __restrict__ h,
                             const float* __restrict__ att_s,
                             const float* __restrict__ att_d,
                             float* __restrict__ as, float* __restrict__ ad,
                             int N, int H, int C) {
  int t = blockIdx.x * blockDim.x + threadIdx.x;
  if (t >= N * H) return;
  int n = t / H, hh = t % H;
  const float* row = h + (size_t)n * H * C + (size_t)hh * C;
  float ss = 0.f, sd = 0.f;
  for (int c = 0; c < C; ++c) {
    float v = row[c];
    ss += v * att_s[hh * C + c];
    sd += v * att_d[hh * C + c];
  }
  as[t] = ss; ad[t] = sd;
}

// ---------- edge kernels (segment softmax) ----------
template<int H>
__global__ void edge_max_kernel(const int* __restrict__ ei, int E, int N,
                                const float* __restrict__ as, const float* __restrict__ ad,
                                unsigned* __restrict__ nmax) {
  int e = blockIdx.x * blockDim.x + threadIdx.x;
  if (e >= E + N) return;
  int s = (e < E) ? ei[e] : (e - E);
  int d = (e < E) ? ei[E + e] : (e - E);
  #pragma unroll
  for (int h = 0; h < H; ++h) {
    float v = lrelu(as[s * H + h] + ad[d * H + h]);
    atomicMax(&nmax[d * H + h], enc_f32(v));
  }
}

template<int H>
__global__ void edge_exp_kernel(const int* __restrict__ ei, int E, int N,
                                const float* __restrict__ as, const float* __restrict__ ad,
                                const unsigned* __restrict__ nmax,
                                float* __restrict__ denom, float* __restrict__ p) {
  int e = blockIdx.x * blockDim.x + threadIdx.x;
  if (e >= E + N) return;
  int s = (e < E) ? ei[e] : (e - E);
  int d = (e < E) ? ei[E + e] : (e - E);
  #pragma unroll
  for (int h = 0; h < H; ++h) {
    float v  = lrelu(as[s * H + h] + ad[d * H + h]);
    float pe = expf(v - dec_f32(nmax[d * H + h]));
    p[e * H + h] = pe;
    atomicAdd(&denom[d * H + h], pe);
  }
}

// ---------- weighted aggregation: out[dst, c] += feat[src, c] * alpha ----------
template<int HC, int C>
__global__ void aggregate_kernel(const int* __restrict__ ei, int E, int N,
                                 const float* __restrict__ feat,
                                 const float* __restrict__ p,
                                 const float* __restrict__ denom,
                                 float* __restrict__ out) {
  int t = blockIdx.x * blockDim.x + threadIdx.x;
  int total = (E + N) * HC;               // <= ~109M, fits int
  if (t >= total) return;
  int e = t / HC, c = t % HC;
  constexpr int H = HC / C;
  int h = c / C;
  int s = (e < E) ? ei[e] : (e - E);
  int d = (e < E) ? ei[E + e] : (e - E);
  float alpha = p[e * H + h] / (denom[d * H + h] + 1e-16f);
  atomicAdd(&out[(size_t)d * HC + c], feat[(size_t)s * HC + c] * alpha);
}

// ---------- final log-softmax over NCLS, in place ----------
__global__ void log_softmax_kernel(float* __restrict__ out, int N) {
  int n = blockIdx.x * blockDim.x + threadIdx.x;
  if (n >= N) return;
  float v[GAT_NCLS];
  float m = -INFINITY;
  #pragma unroll
  for (int c = 0; c < GAT_NCLS; ++c) { v[c] = out[(size_t)n * GAT_NCLS + c]; m = fmaxf(m, v[c]); }
  float s = 0.f;
  #pragma unroll
  for (int c = 0; c < GAT_NCLS; ++c) s += expf(v[c] - m);
  float l = logf(s);
  #pragma unroll
  for (int c = 0; c < GAT_NCLS; ++c) out[(size_t)n * GAT_NCLS + c] = v[c] - m - l;
}

// ---------- host launcher ----------
extern "C" void kernel_launch(void* const* d_in, const int* in_sizes, int n_in,
                              void* d_out, int out_size, void* d_ws, size_t ws_size,
                              hipStream_t stream) {
  const float* x      = (const float*)d_in[0];
  const int*   ei     = (const int*)  d_in[1];
  // d_in[2] edge_attr: unused by the model (edge_dim=None)
  const float* W1     = (const float*)d_in[3];
  const float* att_s1 = (const float*)d_in[4];
  const float* att_d1 = (const float*)d_in[5];
  const float* b1     = (const float*)d_in[6];
  const float* W2     = (const float*)d_in[7];
  const float* att_s2 = (const float*)d_in[8];
  const float* att_d2 = (const float*)d_in[9];
  const float* b2     = (const float*)d_in[10];
  float* out = (float*)d_out;

  const int N    = in_sizes[0] / GAT_FIN;
  const int E    = in_sizes[1] / 2;
  const int Etot = E + N;

  // carve workspace
  char* base = (char*)d_ws;
  size_t off = 0;
  auto carve = [&](size_t bytes) -> void* {
    void* pp = base + off;
    off += bytes;
    off = (off + 255) & ~(size_t)255;
    return pp;
  };
  unsigned short* xb    = (unsigned short*)carve((size_t)N * 128 * 2);
  unsigned short* w1sw  = (unsigned short*)carve((size_t)128 * 128 * 2);
  unsigned short* w2sw  = (unsigned short*)carve((size_t)128 * 16 * 2);
  float*    h1     = (float*)   carve((size_t)N * 128 * 4);
  float*    as1    = (float*)   carve((size_t)N * 2 * 4);
  float*    ad1    = (float*)   carve((size_t)N * 2 * 4);
  unsigned* nmax1  = (unsigned*)carve((size_t)N * 2 * 4);
  float*    denom1 = (float*)   carve((size_t)N * 2 * 4);
  float*    p1     = (float*)   carve((size_t)Etot * 2 * 4);
  float*    agg1   = (float*)   carve((size_t)N * 128 * 4);
  unsigned short* actb = (unsigned short*)carve((size_t)N * 128 * 2);
  float*    h2     = (float*)   carve((size_t)N * 16 * 4);
  float*    as2    = (float*)   carve((size_t)N * 4);
  float*    ad2    = (float*)   carve((size_t)N * 4);
  unsigned* nmax2  = (unsigned*)carve((size_t)N * 4);
  float*    denom2 = (float*)   carve((size_t)N * 4);
  float*    p2     = (float*)   carve((size_t)Etot * 4);

  const int T = 256;
  auto nb = [](long long n, int t) { return (unsigned)((n + t - 1) / t); };

  // ---- layer 1 ----
  f32_to_bf16_kernel<<<nb((long long)N * 128, T), T, 0, stream>>>(x, xb, N * 128);
  swizzle_B_kernel<128><<<nb(128 * 128, T), T, 0, stream>>>(W1, w1sw);
  swizzle_B_kernel<16><<<nb(128 * 16, T), T, 0, stream>>>(W2, w2sw);

  gemm_bf16_wmma<128><<<nb(N, 128), 256, 0, stream>>>(xb, w1sw, h1, N);

  alpha_kernel<<<nb((long long)N * 2, T), T, 0, stream>>>(h1, att_s1, att_d1, as1, ad1, N, 2, 64);
  fill_u32_kernel<<<nb((long long)N * 2, T), T, 0, stream>>>(nmax1, 0u, N * 2);
  fill_u32_kernel<<<nb((long long)N * 2, T), T, 0, stream>>>((unsigned*)denom1, 0u, N * 2);
  edge_max_kernel<2><<<nb(Etot, T), T, 0, stream>>>(ei, E, N, as1, ad1, nmax1);
  edge_exp_kernel<2><<<nb(Etot, T), T, 0, stream>>>(ei, E, N, as1, ad1, nmax1, denom1, p1);
  bias_rows_kernel<<<nb((long long)N * 128, T), T, 0, stream>>>(agg1, b1, N * 128, 128);
  aggregate_kernel<128, 64><<<nb((long long)Etot * 128, T), T, 0, stream>>>(
      ei, E, N, h1, p1, denom1, agg1);
  elu_to_bf16_kernel<<<nb((long long)N * 128, T), T, 0, stream>>>(agg1, actb, N * 128);

  // ---- layer 2 ----
  gemm_bf16_wmma<16><<<nb(N, 128), 256, 0, stream>>>(actb, w2sw, h2, N);

  alpha_kernel<<<nb(N, T), T, 0, stream>>>(h2, att_s2, att_d2, as2, ad2, N, 1, 16);
  fill_u32_kernel<<<nb(N, T), T, 0, stream>>>(nmax2, 0u, N);
  fill_u32_kernel<<<nb(N, T), T, 0, stream>>>((unsigned*)denom2, 0u, N);
  edge_max_kernel<1><<<nb(Etot, T), T, 0, stream>>>(ei, E, N, as2, ad2, nmax2);
  edge_exp_kernel<1><<<nb(Etot, T), T, 0, stream>>>(ei, E, N, as2, ad2, nmax2, denom2, p2);
  bias_rows_kernel<<<nb((long long)N * 16, T), T, 0, stream>>>(out, b2, N * 16, 16);
  aggregate_kernel<16, 16><<<nb((long long)Etot * 16, T), T, 0, stream>>>(
      ei, E, N, h2, p2, denom2, out);

  log_softmax_kernel<<<nb(N, T), T, 0, stream>>>(out, N);

  (void)n_in; (void)out_size; (void)ws_size;
}
